// GPSPlusPlus_59803124630037
// MI455X (gfx1250) — matrix operand order repaired and checked by reading, not compiled, not run on hardware
//
#include <hip/hip_runtime.h>
#include <math.h>

#define DM 512
#define NH 16
#define HD 32
#define NB 64
#define MX 128

typedef _Float16 h16;
typedef __attribute__((ext_vector_type(16))) _Float16 v16h;
typedef __attribute__((ext_vector_type(8)))  _Float16 v8h;
typedef __attribute__((ext_vector_type(8)))  float    v8f;

union V16U { v16h v; v8h h[2]; };

// A fragment (16x32 f16) from row-major [M][K]: lane m=L&15, half h=L>>4
// elements: k = k0 + 8h + e%8  (e<8),  k0 + 16 + 8h + e%8 (e>=8)
static __device__ __forceinline__ v16h load_a16(const h16* __restrict__ row, int k0, int half) {
  V16U u;
  u.h[0] = *(const v8h*)(row + k0 + 8 * half);
  u.h[1] = *(const v8h*)(row + k0 + 16 + 8 * half);
  return u.v;
}
// B fragment (32x16 f16) from row-major [N][K] (i.e. B^T): lane n=L&15, half h
// elements: k = k0 + 8h + e%8  (e<8),  k0 + 16 + 8h + e%8 (e>=8)
static __device__ __forceinline__ v16h load_b16(const h16* __restrict__ colrow, int k0, int half) {
  V16U u;
  u.h[0] = *(const v8h*)(colrow + k0 + 8 * half);
  u.h[1] = *(const v8h*)(colrow + k0 + 16 + 8 * half);
  return u.v;
}
static __device__ __forceinline__ v8f wmma16(v16h a, v16h b, v8f c) {
  return __builtin_amdgcn_wmma_f32_16x16x32_f16(false, a, false, b, (short)0, c, false, false);
}

// ---------- tiny prep kernels ----------
__global__ void gps_starts(const int* __restrict__ batch, int* __restrict__ starts, int ntot) {
  int i = blockIdx.x * 256 + threadIdx.x;
  if (i < ntot) {
    int b = batch[i];
    if (i == 0) starts[0] = 0;
    else if (batch[i - 1] != b) starts[b] = i;
    if (i == ntot - 1) starts[NB] = ntot;
  }
}

__global__ void gps_rowmap(const int* __restrict__ batch, const int* __restrict__ starts,
                           int* __restrict__ rowmap, int ntot) {
  int r = blockIdx.x * 256 + threadIdx.x;
  if (r < ntot) {
    int b = batch[r];
    rowmap[r] = (b << 7) + (r - starts[b]);
  }
}

__global__ void gps_cast(const float* __restrict__ in, h16* __restrict__ out, int n) {
  int i = (blockIdx.x * 256 + threadIdx.x) * 4;
  if (i + 3 < n) {
    float4 f = *(const float4*)(in + i);
    out[i + 0] = (h16)f.x; out[i + 1] = (h16)f.y;
    out[i + 2] = (h16)f.z; out[i + 3] = (h16)f.w;
  } else {
    for (int j = i; j < n; ++j) out[j] = (h16)in[j];
  }
}

__global__ void gps_zero16(uint4* __restrict__ p, int n16) {
  int i = blockIdx.x * 256 + threadIdx.x;
  if (i < n16) p[i] = make_uint4(0u, 0u, 0u, 0u);
}

// ---------- QKV projection: out = x @ W^T + b, scattered into padded head layout ----------
// vmode==0: out[((b*NH+h)*MX + pos)*HD + d]   (Q, K)
// vmode==1: out[((b*NH+h)*HD + d)*MX + pos]   (V transposed per head)
__global__ void gps_qkv(const h16* __restrict__ X, const h16* __restrict__ W,
                        const float* __restrict__ bias, const int* __restrict__ rowmap,
                        h16* __restrict__ out, int ntot, int vmode) {
  int wave = threadIdx.x >> 5;
  int lane = threadIdx.x & 31;
  int lo16 = lane & 15, hh = lane >> 4;
  int m0 = blockIdx.x * 128 + wave * 16;
  int n0 = blockIdx.y * 64;
  if (m0 >= ntot) return;

  int arow = m0 + lo16; if (arow >= ntot) arow = ntot - 1;
  const h16* aptr = X + (size_t)arow * DM;

  v8f acc[4] = {};
  for (int k0 = 0; k0 < DM; k0 += 32) {
    v16h a = load_a16(aptr, k0, hh);
#pragma unroll
    for (int t = 0; t < 4; ++t) {
      v16h b = load_b16(W + (size_t)(n0 + t * 16 + lo16) * DM, k0, hh);
      acc[t] = wmma16(a, b, acc[t]);
    }
  }
#pragma unroll
  for (int g = 0; g < 8; ++g) {
    int r = m0 + g + 8 * hh;
    if (r >= ntot) continue;
    int dense = rowmap[r];
    int b = dense >> 7, pos = dense & (MX - 1);
#pragma unroll
    for (int t = 0; t < 4; ++t) {
      int n = n0 + t * 16 + lo16;
      float v = acc[t][g] + bias[n];
      int head = n >> 5, d = n & (HD - 1);
      size_t idx;
      if (vmode) idx = ((size_t)(b * NH + head) * HD + d) * MX + pos;
      else       idx = ((size_t)(b * NH + head) * MX + pos) * HD + d;
      out[idx] = (h16)v;
    }
  }
}

// ---------- attention: one block per (b,h), 8 waves x 16 q-rows ----------
__global__ void gps_attn(const h16* __restrict__ Q, const h16* __restrict__ K,
                         const h16* __restrict__ Vt, const float* __restrict__ bias,
                         h16* __restrict__ Od, const int* __restrict__ starts, float scale) {
  __shared__ h16 plds[8 * 16 * MX];  // 32 KB: per-wave 16x128 prob tile
  int bh = blockIdx.x;
  int b = bh >> 4, head = bh & 15;
  int wave = threadIdx.x >> 5;
  int lane = threadIdx.x & 31;
  int lo16 = lane & 15, hh = lane >> 4;
  int cnt = starts[b + 1] - starts[b];

  const h16*  qb = Q  + (size_t)bh * MX * HD;
  const h16*  kb = K  + (size_t)bh * MX * HD;
  const h16*  vb = Vt + (size_t)bh * HD * MX;
  const float* bb = bias + (size_t)bh * MX * MX;
  int m0 = wave * 16;

  __builtin_prefetch(bb + (size_t)(m0 + lo16) * MX, 0, 1);

  // S = Q K^T : 8 tiles across the 128 key columns, single K-step (HS=32)
  v8f accS[8] = {};
  {
    v16h a = load_a16(qb + (size_t)(m0 + lo16) * HD, 0, hh);
#pragma unroll
    for (int t = 0; t < 8; ++t) {
      v16h bf = load_b16(kb + (size_t)(t * 16 + lo16) * HD, 0, hh);
      accS[t] = wmma16(a, bf, accS[t]);
    }
  }

  // scale + bias + key mask
  float s[8][8];
#pragma unroll
  for (int t = 0; t < 8; ++t)
#pragma unroll
    for (int g = 0; g < 8; ++g) {
      int m = m0 + g + 8 * hh;
      int n = t * 16 + lo16;
      float v = accS[t][g] * scale + bb[(size_t)m * MX + n];
      s[t][g] = (n < cnt) ? v : -__builtin_inff();
    }

  // softmax across 128 keys: 8 regs per lane + 16-lane half-group reduction
#pragma unroll
  for (int g = 0; g < 8; ++g) {
    float mx = s[0][g];
#pragma unroll
    for (int t = 1; t < 8; ++t) mx = fmaxf(mx, s[t][g]);
#pragma unroll
    for (int off = 1; off < 16; off <<= 1) mx = fmaxf(mx, __shfl_xor(mx, off, 32));
    float sum = 0.0f;
#pragma unroll
    for (int t = 0; t < 8; ++t) { float e = __expf(s[t][g] - mx); s[t][g] = e; sum += e; }
#pragma unroll
    for (int off = 1; off < 16; off <<= 1) sum += __shfl_xor(sum, off, 32);
    float inv = 1.0f / sum;
#pragma unroll
    for (int t = 0; t < 8; ++t) s[t][g] *= inv;
  }

  // stage probabilities as f16 in LDS (row-major 16x128 per wave)
  h16* pw = plds + wave * 16 * MX;
#pragma unroll
  for (int t = 0; t < 8; ++t)
#pragma unroll
    for (int g = 0; g < 8; ++g)
      pw[(g + 8 * hh) * MX + t * 16 + lo16] = (h16)s[t][g];
  __syncthreads();

  // O = P @ V : K=128 in 4 steps, 2 output tiles (HS=32)
  v8f accO[2] = {};
  for (int k0 = 0; k0 < MX; k0 += 32) {
    v16h a = load_a16(pw + (size_t)lo16 * MX, k0, hh);  // ds_load fragments
#pragma unroll
    for (int t2 = 0; t2 < 2; ++t2) {
      v16h bf = load_b16(vb + (size_t)(t2 * 16 + lo16) * MX, k0, hh);
      accO[t2] = wmma16(a, bf, accO[t2]);
    }
  }
  // write dense O: [b][128][512] with col = head*32 + d
#pragma unroll
  for (int t2 = 0; t2 < 2; ++t2)
#pragma unroll
    for (int g = 0; g < 8; ++g) {
      int m = m0 + g + 8 * hh;
      int d = t2 * 16 + lo16;
      Od[((size_t)b * MX + m) * DM + head * HD + d] = (h16)accO[t2][g];
    }
}

// ---------- output projection + bias + residual ----------
__global__ void gps_proj(const h16* __restrict__ Od, const h16* __restrict__ W,
                         const float* __restrict__ bias, const float* __restrict__ Xres,
                         const int* __restrict__ rowmap, float* __restrict__ out, int ntot) {
  int wave = threadIdx.x >> 5;
  int lane = threadIdx.x & 31;
  int lo16 = lane & 15, hh = lane >> 4;
  int m0 = blockIdx.x * 128 + wave * 16;
  int n0 = blockIdx.y * 64;
  if (m0 >= ntot) return;

  int arow = m0 + lo16; if (arow >= ntot) arow = ntot - 1;
  const h16* aptr = Od + (size_t)rowmap[arow] * DM;

  v8f acc[4] = {};
  for (int k0 = 0; k0 < DM; k0 += 32) {
    v16h a = load_a16(aptr, k0, hh);
#pragma unroll
    for (int t = 0; t < 4; ++t) {
      v16h b = load_b16(W + (size_t)(n0 + t * 16 + lo16) * DM, k0, hh);
      acc[t] = wmma16(a, b, acc[t]);
    }
  }
#pragma unroll
  for (int g = 0; g < 8; ++g) {
    int r = m0 + g + 8 * hh;
    if (r >= ntot) continue;
#pragma unroll
    for (int t = 0; t < 4; ++t) {
      int n = n0 + t * 16 + lo16;
      out[(size_t)r * DM + n] = acc[t][g] + bias[n] + Xres[(size_t)r * DM + n];
    }
  }
}

// ---------- LayerNorm in-place, one wave per row ----------
__global__ void gps_ln(float* __restrict__ io, const float* __restrict__ w,
                       const float* __restrict__ bsh, int ntot) {
  int wave = threadIdx.x >> 5;
  int lane = threadIdx.x & 31;
  int r = blockIdx.x * 8 + wave;
  if (r >= ntot) return;
  float* row = io + (size_t)r * DM;
  float v[16], sm = 0.0f, sq = 0.0f;
#pragma unroll
  for (int i = 0; i < 16; ++i) {
    v[i] = row[lane + 32 * i];
    sm += v[i]; sq += v[i] * v[i];
  }
#pragma unroll
  for (int off = 1; off < 32; off <<= 1) {
    sm += __shfl_xor(sm, off, 32);
    sq += __shfl_xor(sq, off, 32);
  }
  float mean = sm * (1.0f / DM);
  float var  = sq * (1.0f / DM) - mean * mean;
  float rstd = rsqrtf(var + 1e-5f);
#pragma unroll
  for (int i = 0; i < 16; ++i) {
    int c = lane + 32 * i;
    row[c] = (v[i] - mean) * rstd * w[c] + bsh[c];
  }
}

extern "C" void kernel_launch(void* const* d_in, const int* in_sizes, int n_in,
                              void* d_out, int out_size, void* d_ws, size_t ws_size,
                              hipStream_t stream) {
  const float* x   = (const float*)d_in[0];
  const float* ab  = (const float*)d_in[1];
  const float* Wq  = (const float*)d_in[2];  const float* bq = (const float*)d_in[3];
  const float* Wk  = (const float*)d_in[4];  const float* bk = (const float*)d_in[5];
  const float* Wv  = (const float*)d_in[6];  const float* bv = (const float*)d_in[7];
  const float* Wp  = (const float*)d_in[8];  const float* bp = (const float*)d_in[9];
  const float* lnw = (const float*)d_in[10]; const float* lnb = (const float*)d_in[11];
  const int*   batch = (const int*)d_in[12];
  int ntot = in_sizes[0] / DM;

  size_t off = 0;
  auto wsalloc = [&](size_t bytes) -> void* {
    void* p = (char*)d_ws + off;
    off += (bytes + 255) & ~(size_t)255;
    return p;
  };
  h16* xh  = (h16*)wsalloc((size_t)ntot * DM * 2);
  h16* wqh = (h16*)wsalloc((size_t)DM * DM * 2);
  h16* wkh = (h16*)wsalloc((size_t)DM * DM * 2);
  h16* wvh = (h16*)wsalloc((size_t)DM * DM * 2);
  h16* wph = (h16*)wsalloc((size_t)DM * DM * 2);
  h16* qws = (h16*)wsalloc((size_t)NB * NH * MX * HD * 2);
  h16* kws = (h16*)wsalloc((size_t)NB * NH * MX * HD * 2);
  h16* vws = (h16*)wsalloc((size_t)NB * NH * HD * MX * 2);
  h16* od  = (h16*)wsalloc((size_t)NB * MX * DM * 2);
  int* rowmap  = (int*)wsalloc((size_t)ntot * 4);
  int* dstarts = (int*)wsalloc(65 * 4);

  const int tb = 256;
  int nblk = (ntot + tb - 1) / tb;
  gps_starts<<<nblk, tb, 0, stream>>>(batch, dstarts, ntot);
  gps_rowmap<<<nblk, tb, 0, stream>>>(batch, dstarts, rowmap, ntot);

  int nx = ntot * DM;
  gps_cast<<<((nx + 3) / 4 + tb - 1) / tb, tb, 0, stream>>>(x, xh, nx);
  int nw = DM * DM;
  int cwb = ((nw + 3) / 4 + tb - 1) / tb;
  gps_cast<<<cwb, tb, 0, stream>>>(Wq, wqh, nw);
  gps_cast<<<cwb, tb, 0, stream>>>(Wk, wkh, nw);
  gps_cast<<<cwb, tb, 0, stream>>>(Wv, wvh, nw);
  gps_cast<<<cwb, tb, 0, stream>>>(Wp, wph, nw);

  int vz16 = (NB * NH * HD * MX * 2) / 16;  // uint4 count
  gps_zero16<<<(vz16 + tb - 1) / tb, tb, 0, stream>>>((uint4*)vws, vz16);

  dim3 gq((ntot + 127) / 128, DM / 64);
  gps_qkv<<<gq, 256, 0, stream>>>(xh, wqh, bq, rowmap, qws, ntot, 0);
  gps_qkv<<<gq, 256, 0, stream>>>(xh, wkh, bk, rowmap, kws, ntot, 0);
  gps_qkv<<<gq, 256, 0, stream>>>(xh, wvh, bv, rowmap, vws, ntot, 1);

  gps_attn<<<NB * NH, 256, 0, stream>>>(qws, kws, vws, ab, od, dstarts,
                                        0.17677669529663687f /* 1/sqrt(32) */);

  gps_proj<<<gq, 256, 0, stream>>>(od, wph, bp, x, rowmap, (float*)d_out, ntot);

  gps_ln<<<(ntot + 7) / 8, 256, 0, stream>>>((float*)d_out, lnw, lnb, ntot);
}
